// StyleConditionedSSGO_17188459119281
// MI455X (gfx1250) — matrix operand order repaired and checked by reading, not compiled, not run on hardware
//
#include <hip/hip_runtime.h>
#include <math.h>

// ---------------------------------------------------------------------------
// StyleConditionedSSGO forward for MI455X (gfx1250).
// Per-voxel channel-mixing GEMMs (262144 x 48 @ 48 x 48) use
// v_wmma_f32_16x16x32_f16 (K padded to 64), templated epilogues, vectorized
// b128 staging. Spectral parts use twiddle-recurrence truncated DFTs (M=6).
// All transcendentals are branchless (v_exp / v_rcp) to keep EXEC uniform.
// ---------------------------------------------------------------------------

typedef _Float16 h16 __attribute__((ext_vector_type(16)));
typedef _Float16 h8  __attribute__((ext_vector_type(8)));
typedef float    f8v __attribute__((ext_vector_type(8)));

#define NV   262144      // 64^3 voxels
#define NC   48
#define PI_F 3.14159265358979f

__device__ __forceinline__ float rcp_f(float x) { return __builtin_amdgcn_rcpf(x); }
__device__ __forceinline__ float tanh_fast(float x) {
  return 1.f - 2.f * rcp_f(__expf(2.f * x) + 1.f);   // branchless, saturates
}
__device__ __forceinline__ float gelu_f(float x) {
  float x3 = x * x * x;
  return 0.5f * x * (1.f + tanh_fast(0.7978845608028654f * (x + 0.044715f * x3)));
}
__device__ __forceinline__ float sigm_f(float x) { return rcp_f(1.f + __expf(-x)); }

// ---------------------------------------------------------------------------
// WMMA GEMM: out[r][n] = epi( accIn[r][n] + bias[n] + sum_k X[r][k]*W[k][n] )
// ACT: 0 = none, 1 = gelu, 2 = gelu * (occ[row] > 0)
// X: (nrows x 48) f32 row-major, W: (48 x 48) f32 row-major, nrows % 128 == 0.
// Block = 256 thr = 8 waves; wave owns a 16-row slab; block does 128 rows.
// ---------------------------------------------------------------------------
template<int ACT, bool HASACC>
__global__ __launch_bounds__(256) void k_gemm48(
    const float* __restrict__ X, const float* __restrict__ W,
    const float* __restrict__ bias, const float* __restrict__ accIn,
    float* __restrict__ out, const float* __restrict__ occ)
{
  __shared__ _Float16 Wt[NC * 64];    // W transposed: Wt[n*64 + k], K padded w/ 0
  __shared__ _Float16 Xh[128 * 64];   // X tile f16:  Xh[r*64 + k]

  const int tid  = threadIdx.x;
  const int lane = tid & 31;
  const int wave = tid >> 5;
  const int r0   = blockIdx.x * 128;

  __builtin_prefetch(&X[(size_t)r0 * NC], 0, 1);   // global_prefetch_b8

  // ---- stage W (2304 f32 = 576 float4), transpose into LDS ----
  for (int i = tid; i < 576; i += 256) {
    int k = i / 12, n4 = (i % 12) * 4;
    float4 w = *(const float4*)&W[k * NC + n4];
    Wt[(n4 + 0) * 64 + k] = (_Float16)w.x;
    Wt[(n4 + 1) * 64 + k] = (_Float16)w.y;
    Wt[(n4 + 2) * 64 + k] = (_Float16)w.z;
    Wt[(n4 + 3) * 64 + k] = (_Float16)w.w;
  }
  for (int i = tid; i < NC * 16; i += 256) {       // K padding, 768 = 3*256
    int n = i >> 4, k = 48 + (i & 15);
    Wt[n * 64 + k] = (_Float16)0.f;
  }
  // ---- stage X tile (128 rows * 12 float4 = 1536, uniform 6/thread) ----
  for (int i = tid; i < 1536; i += 256) {
    int r = i / 12, k4 = (i % 12) * 4;
    float4 v = *(const float4*)&X[(size_t)(r0 + r) * NC + k4];
    _Float16* d = &Xh[r * 64 + k4];
    d[0] = (_Float16)v.x; d[1] = (_Float16)v.y;
    d[2] = (_Float16)v.z; d[3] = (_Float16)v.w;
  }
  for (int i = tid; i < 128 * 16; i += 256) {      // K padding, 2048 = 8*256
    int r = i >> 4, k = 48 + (i & 15);
    Xh[r * 64 + k] = (_Float16)0.f;
  }
  __syncthreads();

  const int m0   = wave * 16;
  const int mrow = m0 + (lane & 15);

  // A fragments (16-bit A 16x32): lane<16 -> K base 0, else 16;
  // halves 0..15 = K kb..kb+15 (contiguous, matches B). Two b128 loads.
  const int kb = (lane < 16) ? 0 : 16;
  h8 lo0 = *(const h8*)&Xh[mrow * 64 + kb];
  h8 hi0 = *(const h8*)&Xh[mrow * 64 + kb + 8];
  h8 lo1 = *(const h8*)&Xh[mrow * 64 + 32 + kb];
  h8 hi1 = *(const h8*)&Xh[mrow * 64 + 40 + kb];
  h16 a0 = __builtin_shufflevector(lo0, hi0, 0,1,2,3,4,5,6,7,8,9,10,11,12,13,14,15);
  h16 a1 = __builtin_shufflevector(lo1, hi1, 0,1,2,3,4,5,6,7,8,9,10,11,12,13,14,15);

#pragma unroll
  for (int t = 0; t < NC / 16; ++t) {
    const int n   = (lane & 15) + 16 * t;
    const int kb2 = (lane < 16) ? 0 : 16;
    // B fragment (32x16): lane = n-col, 16 contiguous K halves.
    h16 b0 = *(const h16*)&Wt[n * 64 + kb2];
    h16 b1 = *(const h16*)&Wt[n * 64 + 32 + kb2];

    const int rbase = r0 + m0 + ((lane < 16) ? 0 : 8);
    const int col   = 16 * t + (lane & 15);

    f8v c;
#pragma unroll
    for (int v = 0; v < 8; ++v) c[v] = bias[col];
    if (HASACC) {
#pragma unroll
      for (int v = 0; v < 8; ++v) c[v] += accIn[(size_t)(rbase + v) * NC + col];
    }
    c = __builtin_amdgcn_wmma_f32_16x16x32_f16(false, a0, false, b0, (short)0, c, false, false);
    c = __builtin_amdgcn_wmma_f32_16x16x32_f16(false, a1, false, b1, (short)0, c, false, false);
#pragma unroll
    for (int v = 0; v < 8; ++v) {
      float y = c[v];
      if (ACT >= 1) y = gelu_f(y);
      if (ACT == 2) y *= (occ[rbase + v] > 0.f) ? 1.f : 0.f;
      out[(size_t)(rbase + v) * NC + col] = y;
    }
  }
}

// ------------------------- small dense helpers ------------------------------
// out[j] = act( sum_i v[i]*W[i*n+j] + b[j] ),  1-row matvec
__global__ void k_vecmat(const float* __restrict__ v, const float* __restrict__ W,
                         const float* __restrict__ b, float* __restrict__ out,
                         int k, int n, int act)
{
  for (int j = blockIdx.x * blockDim.x + threadIdx.x; j < n; j += gridDim.x * blockDim.x) {
    float s = b ? b[j] : 0.f;
    for (int i = 0; i < k; ++i) s += v[i] * W[i * n + j];
    out[j] = (act == 1) ? gelu_f(s) : s;
  }
}

// 11 fused 128->n matvecs (CP factors + mode deltas); block b = matvec b.
struct FacArgs { const float* W[11]; const float* B[11]; float* O[11]; int n[11]; };
__global__ void k_facs(const float* __restrict__ hf, FacArgs a)
{
  int b = blockIdx.x;
  int n = a.n[b];
  const float* W  = a.W[b];
  const float* bb = a.B[b];
  float* o        = a.O[b];
  for (int j = threadIdx.x; j < n; j += blockDim.x) {
    float s = bb[j];
    for (int i = 0; i < 128; ++i) s += hf[i] * W[i * n + j];
    o[j] = s;
  }
}

// ------------------------- truncated DFT stages -----------------------------
// forward: in[pre][64][post] -> out[pre][6][post],  tw = e^{-2pi i k x / 64}
// one sincosf per thread; per-step rotation recurrence.
__global__ void k_dft_f(const float* __restrict__ inR, const float* __restrict__ inI,
                        float* __restrict__ outR, float* __restrict__ outI,
                        int pre, int post)
{
  long total = (long)pre * 6 * post;
  for (long idx = (long)blockIdx.x * blockDim.x + threadIdx.x; idx < total;
       idx += (long)gridDim.x * blockDim.x) {
    int  q  = (int)(idx % post);
    long t  = idx / post;
    int  kk = (int)(t % 6);
    long p  = t / 6;
    const float* bR = inR + (p * 64) * (long)post + q;
    const float* bI = inI ? inI + (p * 64) * (long)post + q : nullptr;
    float snd, csd; sincosf(-(PI_F / 32.f) * (float)kk, &snd, &csd);
    float cs = 1.f, sn = 0.f;
    float sr = 0.f, si = 0.f;
    for (int x = 0; x < 64; ++x) {
      float vr = bR[(long)x * post];
      if (bI) { float vi = bI[(long)x * post];
                sr += vr * cs - vi * sn; si += vr * sn + vi * cs; }
      else    { sr += vr * cs; si += vr * sn; }
      float c2 = cs * csd - sn * snd;
      sn = cs * snd + sn * csd; cs = c2;
    }
    outR[idx] = sr; outI[idx] = si;
  }
}

// inverse (full ifft axis): in[pre][6][post] -> out[pre][64][post], (1/64) e^{+2pi i kx/64}
__global__ void k_idft(const float* __restrict__ inR, const float* __restrict__ inI,
                       float* __restrict__ outR, float* __restrict__ outI,
                       int pre, int post)
{
  long total = (long)pre * 64 * post;
  for (long idx = (long)blockIdx.x * blockDim.x + threadIdx.x; idx < total;
       idx += (long)gridDim.x * blockDim.x) {
    int  q = (int)(idx % post);
    long t = idx / post;
    int  x = (int)(t % 64);
    long p = t / 64;
    float snd, csd; sincosf((PI_F / 32.f) * (float)x, &snd, &csd);
    float cs = 1.f, sn = 0.f;
    float sr = 0.f, si = 0.f;
#pragma unroll
    for (int kk = 0; kk < 6; ++kk) {
      long  ii = (p * 6 + kk) * (long)post + q;
      float vr = inR[ii], vi = inI[ii];
      sr += vr * cs - vi * sn;
      si += vr * sn + vi * cs;
      float c2 = cs * csd - sn * snd;
      sn = cs * snd + sn * csd; cs = c2;
    }
    outR[idx] = sr * (1.f / 64.f);
    outI[idx] = si * (1.f / 64.f);
  }
}

// last-axis irfft with 6 modes: out real [pre][64][post]
__global__ void k_irfft6(const float* __restrict__ inR, const float* __restrict__ inI,
                         float* __restrict__ out, int pre, int post)
{
  long total = (long)pre * 64 * post;
  for (long idx = (long)blockIdx.x * blockDim.x + threadIdx.x; idx < total;
       idx += (long)gridDim.x * blockDim.x) {
    int  q = (int)(idx % post);
    long t = idx / post;
    int  z = (int)(t % 64);
    long p = t / 64;
    long b = p * 6 * (long)post + q;
    float snd, csd; sincosf((PI_F / 32.f) * (float)z, &snd, &csd);
    float cs = csd, sn = snd;          // k = 1 term
    float s = inR[b];                  // k = 0 term (Re only)
#pragma unroll
    for (int kk = 1; kk < 6; ++kk) {
      long ii = b + (long)kk * post;
      s += 2.f * (inR[ii] * cs - inI[ii] * sn);
      float c2 = cs * csd - sn * snd;
      sn = cs * snd + sn * csd; cs = c2;
    }
    out[idx] = s * (1.f / 64.f);
  }
}

// ------------------------- spectral weights & mix ---------------------------
// Weff = base + 0.1 * sum_r a[r,i] b[r,o] cx[r,kx] cy[r,ky] cz[r,kz]
// fac layout: a@0(96) b@96(96) cx@192(12) cy@204(12) cz@216(12)
__global__ void k_build_weff(const float* __restrict__ baseR, const float* __restrict__ baseI,
                             const float* __restrict__ fR, const float* __restrict__ fI,
                             float* __restrict__ WeR, float* __restrict__ WeI)
{
  const int total = 48 * 48 * 216;
  for (int idx = blockIdx.x * blockDim.x + threadIdx.x; idx < total;
       idx += gridDim.x * blockDim.x) {
    int k3 = idx % 216, io = idx / 216, o = io % 48, i = io / 48;
    int kx = k3 / 36, ky = (k3 / 6) % 6, kz = k3 % 6;
    float dr = 0.f, di = 0.f;
#pragma unroll
    for (int r = 0; r < 2; ++r) {
      dr += fR[r * 48 + i] * fR[96 + r * 48 + o] * fR[192 + r * 6 + kx] *
            fR[204 + r * 6 + ky] * fR[216 + r * 6 + kz];
      di += fI[r * 48 + i] * fI[96 + r * 48 + o] * fI[192 + r * 6 + kx] *
            fI[204 + r * 6 + ky] * fI[216 + r * 6 + kz];
    }
    WeR[idx] = baseR[idx] + 0.1f * dr;
    WeI[idx] = baseI[idx] + 0.1f * di;
  }
}

// om[k3,o] = mw(k3) * sum_i Gh[k3,i] * Weff[i,o,k3]   (complex)
__global__ void k_specmix(const float* __restrict__ GhR, const float* __restrict__ GhI,
                          const float* __restrict__ WeR, const float* __restrict__ WeI,
                          const float* __restrict__ mode_base, const float* __restrict__ md,
                          float* __restrict__ omR, float* __restrict__ omI)
{
  const int total = 216 * 48;
  for (int idx = blockIdx.x * blockDim.x + threadIdx.x; idx < total;
       idx += gridDim.x * blockDim.x) {
    int o = idx % 48, k3 = idx / 48;
    float mw = sigm_f(mode_base[k3] + md[k3]);
    float sr = 0.f, si = 0.f;
    for (int i = 0; i < 48; ++i) {
      float gr = GhR[k3 * 48 + i], gi = GhI[k3 * 48 + i];
      int   wi_ = (i * 48 + o) * 216 + k3;
      float wr = WeR[wi_], wim = WeI[wi_];
      sr += gr * wr - gi * wim;
      si += gr * wim + gi * wr;
    }
    omR[k3 * 48 + o] = sr * mw;
    omI[k3 * 48 + o] = si * mw;
  }
}

// ------------------------- pointwise / graph / moe --------------------------
__global__ void k_extract_occ(const float* __restrict__ x, float* __restrict__ occ) {
  for (int v = blockIdx.x * blockDim.x + threadIdx.x; v < NV; v += gridDim.x * blockDim.x)
    occ[v] = x[(size_t)v * 6];
}

__global__ void k_abs216(const float* __restrict__ r, const float* __restrict__ i,
                         float* __restrict__ o) {
  int k = blockIdx.x * blockDim.x + threadIdx.x;
  if (k < 216) o[k] = sqrtf(r[k] * r[k] + i[k] * i[k]);
}

// z[j] = z_geom[j] + softplus(alpha) * style_table[sid][j]
__global__ void k_encode(const float* __restrict__ absoh, const float* __restrict__ encW,
                         const float* __restrict__ encb, const float* __restrict__ styleT,
                         const float* __restrict__ alpha, const int* __restrict__ sid,
                         float* __restrict__ z)
{
  int j = threadIdx.x;
  if (j < 32) {
    float s = encb[j];
    for (int k = 0; k < 216; ++k) s += absoh[k] * encW[k * 32 + j];
    float a = log1pf(__expf(alpha[0]));
    z[j] = s + a * styleT[sid[0] * 32 + j];
  }
}

// lift: out[v][c] = sum_{j<6} x[v][j]*W[j][c] + b[c]
__global__ void k_lift(const float* __restrict__ x, const float* __restrict__ W,
                       const float* __restrict__ b, float* __restrict__ out)
{
  long total = (long)NV * NC;
  for (long idx = (long)blockIdx.x * blockDim.x + threadIdx.x; idx < total;
       idx += (long)gridDim.x * blockDim.x) {
    int c = (int)(idx % NC); long v = idx / NC;
    float s = b[c];
#pragma unroll
    for (int j = 0; j < 6; ++j) s += x[v * 6 + j] * W[j * NC + c];
    out[idx] = s;
  }
}

// masked 6-neighbor mean (periodic roll, matching jnp.roll)
__global__ void k_graph_agg(const float* __restrict__ f, const float* __restrict__ occ,
                            float* __restrict__ agg)
{
  long total = (long)NV * NC;
  for (long idx = (long)blockIdx.x * blockDim.x + threadIdx.x; idx < total;
       idx += (long)gridDim.x * blockDim.x) {
    int c = (int)(idx % NC); int v = (int)(idx / NC);
    int z = v & 63, y = (v >> 6) & 63, x = v >> 12;
    float nbr = 0.f, deg = 0.f;
#pragma unroll
    for (int d = 0; d < 6; ++d) {
      int ax = d >> 1, s = (d & 1) ? 1 : -1;
      int xx = x, yy = y, zz = z;
      if (ax == 0) xx = (x + s + 64) & 63;
      else if (ax == 1) yy = (y + s + 64) & 63;
      else zz = (z + s + 64) & 63;
      int nv = (xx << 12) | (yy << 6) | zz;
      float mk = occ[nv] > 0.f ? 1.f : 0.f;
      nbr += f[(size_t)nv * NC + c] * mk;
      deg += mk;
    }
    agg[idx] = nbr * rcp_f(fmaxf(deg, 1.f));
  }
}

// h = gate*g + (1-gate)*f ; gate = sigmoid([g,f] @ gate_W + b). In-place into g.
__global__ void k_gatemix(float* __restrict__ g, const float* __restrict__ f,
                          const float* __restrict__ gW, const float* __restrict__ gb)
{
  for (int v = blockIdx.x * blockDim.x + threadIdx.x; v < NV; v += gridDim.x * blockDim.x) {
    float s = gb[0];
    size_t b = (size_t)v * NC;
    for (int c = 0; c < NC; ++c) s += g[b + c] * gW[c] + f[b + c] * gW[NC + c];
    float gt = sigm_f(s);
    for (int c = 0; c < NC; ++c) g[b + c] = gt * g[b + c] + (1.f - gt) * f[b + c];
  }
}

// fused MoE: softmax gate + 4 experts (48->32 gelu ->48) + weighted combine.
// No eh[] array: each hidden unit accumulates straight into acc[48].
__global__ void k_moe(const float* __restrict__ h, const float* __restrict__ gW,
                      const float* __restrict__ gb, const float* __restrict__ W1,
                      const float* __restrict__ b1, const float* __restrict__ W2,
                      const float* __restrict__ b2, float* __restrict__ out)
{
  for (int v = blockIdx.x * blockDim.x + threadIdx.x; v < NV; v += gridDim.x * blockDim.x) {
    size_t base = (size_t)v * NC;
    float hv[NC];
#pragma unroll
    for (int c = 0; c < NC; ++c) hv[c] = h[base + c];
    float lg[4];
#pragma unroll
    for (int e = 0; e < 4; ++e) {
      float s = gb[e];
#pragma unroll
      for (int c = 0; c < NC; ++c) s += hv[c] * gW[c * 4 + e];
      lg[e] = s;
    }
    float mx = fmaxf(fmaxf(lg[0], lg[1]), fmaxf(lg[2], lg[3]));
    float den = 0.f;
#pragma unroll
    for (int e = 0; e < 4; ++e) { lg[e] = __expf(lg[e] - mx); den += lg[e]; }
    float inv = rcp_f(den);
    float acc[NC];
#pragma unroll
    for (int o = 0; o < NC; ++o) acc[o] = 0.f;
    for (int e = 0; e < 4; ++e) {
      float ge = lg[e] * inv;
#pragma unroll
      for (int o = 0; o < NC; ++o) acc[o] += ge * b2[e * NC + o];
      for (int hh = 0; hh < 32; ++hh) {
        float s = b1[e * 32 + hh];
#pragma unroll
        for (int c = 0; c < NC; ++c) s += hv[c] * W1[(e * NC + c) * 32 + hh];
        float eg = ge * gelu_f(s);
        const float* w2 = &W2[(e * 32 + hh) * NC];
#pragma unroll
        for (int o = 0; o < NC; ++o) acc[o] += eg * w2[o];
      }
    }
#pragma unroll
    for (int o = 0; o < NC; ++o) out[base + o] = acc[o];
  }
}

// head out: out[v*11 + off + o] = act(s @ oW + ob) * occ[v]
__global__ void k_headout(const float* __restrict__ s, const float* __restrict__ oW,
                          const float* __restrict__ ob, float* __restrict__ out,
                          const float* __restrict__ occ, int off, int od, int act)
{
  long total = (long)NV * od;
  for (long idx = (long)blockIdx.x * blockDim.x + threadIdx.x; idx < total;
       idx += (long)gridDim.x * blockDim.x) {
    int o = (int)(idx % od); long v = idx / od;
    float acc = ob[o];
    for (int c = 0; c < NC; ++c) acc += s[v * NC + c] * oW[c * od + o];
    if (act == 1) acc = tanh_fast(acc);
    out[v * 11 + off + o] = acc * occ[v];
  }
}

// b2p[j] = sdf_h2b[j] + sum_{l<32} style_table[sid][l] * sdf_h2W[48+l][j]
__global__ void k_sdf_bias(const float* __restrict__ W2, const float* __restrict__ b2,
                           const float* __restrict__ styleT, const int* __restrict__ sid,
                           float* __restrict__ b2p)
{
  int j = blockIdx.x * blockDim.x + threadIdx.x;
  if (j < NC) {
    float s = b2[j];
    for (int l = 0; l < 32; ++l) s += styleT[sid[0] * 32 + l] * W2[(48 + l) * NC + j];
    b2p[j] = s;
  }
}

// ---------------------------------------------------------------------------
extern "C" void kernel_launch(void* const* d_in, const int* in_sizes, int n_in,
                              void* d_out, int out_size, void* d_ws, size_t ws_size,
                              hipStream_t stream)
{
  (void)in_sizes; (void)n_in; (void)out_size; (void)ws_size;
  auto in = [&](int i) { return (const float*)d_in[i]; };
  const float* X  = in(0);
  const int* SID  = (const int*)d_in[1];
  float* OUT      = (float*)d_out;

  float* ws = (float*)d_ws;
  size_t big = (size_t)NV * NC;
  float *BUF0 = ws, *BUF1 = ws + big, *BUF2 = ws + 2 * big, *BUF3 = ws + 3 * big;
  float* sm  = ws + 4 * big;
  float* occ = sm;  sm += NV;
  float* A1r = sm;  sm += 1179648;  float* A1i = sm;  sm += 1179648;
  float* A2r = sm;  sm += 110592;   float* A2i = sm;  sm += 110592;
  float* GhR = sm;  sm += 10368;    float* GhI = sm;  sm += 10368;
  float* omR = sm;  sm += 10368;    float* omI = sm;  sm += 10368;
  float* WeR = sm;  sm += 497664;   float* WeI = sm;  sm += 497664;
  float* fac = sm;  sm += 512;      // facR @ 0, facI @ 228
  float* md  = sm;  sm += 216;
  float* aoh = sm;  sm += 216;
  float* zv  = sm;  sm += 32;
  float* hf0 = sm;  sm += 128;
  float* hf  = sm;  sm += 128;
  float* b2p = sm;  sm += 48;

  auto NB = [](long n) { return (unsigned)((n + 255) / 256); };

  // ---- encoder: truncated rfftn of occupancy -> |oh| -> z -> hf ----
  k_extract_occ<<<NB(NV), 256, 0, stream>>>(X, occ);
  k_dft_f<<<NB(6L * 4096), 256, 0, stream>>>(occ, nullptr, A1r, A1i, 1, 4096);
  k_dft_f<<<NB(6L * 6 * 64), 256, 0, stream>>>(A1r, A1i, A2r, A2i, 6, 64);
  k_dft_f<<<NB(216), 256, 0, stream>>>(A2r, A2i, GhR, GhI, 36, 1);
  k_abs216<<<1, 256, 0, stream>>>(GhR, GhI, aoh);
  k_encode<<<1, 32, 0, stream>>>(aoh, in(4), in(5), in(2), in(3), SID, zv);
  k_vecmat<<<2, 64, 0, stream>>>(zv, in(6), in(7), hf0, 32, 128, 1);
  k_vecmat<<<2, 64, 0, stream>>>(hf0, in(8), in(9), hf, 128, 128, 1);

  // ---- FNO block helper (spec base index sb) on gbuf, spat scratch = BUF2 ----
  auto fno = [&](float* gbuf, int sb) {
    // all 10 CP-factor matvecs + mode delta in one launch
    FacArgs fa;
    for (int part = 0; part < 2; ++part) {
      int o = sb + part * 10;
      float* fp = fac + part * 228;
      int base = part * 5;
      const float* Ws[5]  = { in(o + 0), in(o + 2), in(o + 4), in(o + 6), in(o + 8) };
      const float* Bs[5]  = { in(o + 1), in(o + 3), in(o + 5), in(o + 7), in(o + 9) };
      float*       Os[5]  = { fp + 0, fp + 96, fp + 192, fp + 204, fp + 216 };
      int          Ns[5]  = { 96, 96, 12, 12, 12 };
      for (int j = 0; j < 5; ++j) {
        fa.W[base + j] = Ws[j]; fa.B[base + j] = Bs[j];
        fa.O[base + j] = Os[j]; fa.n[base + j] = Ns[j];
      }
    }
    fa.W[10] = in(sb + 20); fa.B[10] = in(sb + 21); fa.O[10] = md; fa.n[10] = 216;
    k_facs<<<11, 256, 0, stream>>>(hf, fa);
    // forward truncated rfftn: x, y, z contractions
    k_dft_f<<<NB(6L * 196608), 256, 0, stream>>>(gbuf, nullptr, A1r, A1i, 1, 196608);
    k_dft_f<<<NB(6L * 6 * 3072), 256, 0, stream>>>(A1r, A1i, A2r, A2i, 6, 3072);
    k_dft_f<<<NB(216L * 48), 256, 0, stream>>>(A2r, A2i, GhR, GhI, 36, 48);
    // effective complex weights + per-mode mix
    k_build_weff<<<NB(48L * 48 * 216), 256, 0, stream>>>(in(sb + 22), in(sb + 23),
                                                         fac, fac + 228, WeR, WeI);
    k_specmix<<<NB(216L * 48), 256, 0, stream>>>(GhR, GhI, WeR, WeI, in(sb + 24), md, omR, omI);
    // sparse irfftn: kx, ky (ifft) then kz (irfft) -> spat in BUF2
    k_idft<<<NB(64L * 1728), 256, 0, stream>>>(omR, omI, A2r, A2i, 1, 1728);
    k_idft<<<NB(64L * 64 * 288), 256, 0, stream>>>(A2r, A2i, A1r, A1i, 64, 288);
    k_irfft6<<<NB((long)NV * 48), 256, 0, stream>>>(A1r, A1i, BUF2, 4096, 48);
    // g = gelu(spat + g @ skip_W + skip_b)   [WMMA]
    k_gemm48<1, true><<<NV / 128, 256, 0, stream>>>(gbuf, in(sb + 25), in(sb + 26),
                                                    BUF2, gbuf, nullptr);
  };

  // ---- graph conv helper on f = BUF1 (param base gb), gelu*mask fused ----
  auto graph = [&](int gb) {
    k_graph_agg<<<NB((long)NV * NC), 256, 0, stream>>>(BUF1, occ, BUF2);
    k_gemm48<0, false><<<NV / 128, 256, 0, stream>>>(BUF1, in(gb + 0), in(gb + 1),
                                                     nullptr, BUF3, nullptr);
    k_gemm48<2, true><<<NV / 128, 256, 0, stream>>>(BUF2, in(gb + 2), in(gb + 3),
                                                    BUF3, BUF1, occ);
  };

  // ---- main pipeline ----
  k_lift<<<NB((long)NV * NC), 256, 0, stream>>>(X, in(10), in(11), BUF0);   // g
  fno(BUF0, 40); fno(BUF0, 67); fno(BUF0, 94);
  k_lift<<<NB((long)NV * NC), 256, 0, stream>>>(X, in(12), in(13), BUF1);   // f
  graph(175); graph(179);
  k_gatemix<<<NB(NV), 256, 0, stream>>>(BUF0, BUF1, in(14), in(15));        // h in BUF0
  fno(BUF0, 121); fno(BUF0, 148);
  k_moe<<<NB(NV), 256, 0, stream>>>(BUF0, in(16), in(17), in(18), in(19), in(20), in(21), BUF1);

  // ---- heads (WMMA hidden layers) ----
  auto head = [&](int hw, int hb, int ow, int ob, int off, int od) {
    k_gemm48<1, false><<<NV / 128, 256, 0, stream>>>(BUF1, in(hw), in(hb),
                                                     nullptr, BUF2, nullptr);
    k_headout<<<NB((long)NV * od), 256, 0, stream>>>(BUF2, in(ow), in(ob), OUT, occ, off, od, 0);
  };
  head(28, 29, 30, 31, 0, 6);   // stress
  head(32, 33, 34, 35, 6, 3);   // disp
  head(36, 37, 38, 39, 9, 1);   // phi
  // style sdf: z_style concat folded into a precomputed bias (B=1)
  k_gemm48<1, false><<<NV / 128, 256, 0, stream>>>(BUF1, in(22), in(23), nullptr, BUF2, nullptr);
  k_sdf_bias<<<1, 64, 0, stream>>>(in(24), in(25), in(2), SID, b2p);
  k_gemm48<1, false><<<NV / 128, 256, 0, stream>>>(BUF2, in(24), b2p, nullptr, BUF3, nullptr);
  k_headout<<<NB(NV), 256, 0, stream>>>(BUF3, in(26), in(27), OUT, occ, 10, 1, 1);
}
